// MultiHeadAttention_26328149524719
// MI455X (gfx1250) — matrix operand order repaired and checked by reading, not compiled, not run on hardware
//
#include <hip/hip_runtime.h>
#include <hip/hip_bf16.h>

// MI455X / gfx1250 multi-head attention (with reference's reshape quirks).
// All matmuls via v_wmma_f32_16x16x32_f16, double-buffered LDS staging,
// async global->LDS copies where the toolchain exposes them.

typedef __attribute__((ext_vector_type(16))) _Float16 v16h;
typedef __attribute__((ext_vector_type(8)))  _Float16 v8h;
typedef __attribute__((ext_vector_type(8)))  float    v8f;
typedef __attribute__((ext_vector_type(4)))  float    v4f;
typedef __attribute__((ext_vector_type(4)))  int      v4i;

#define BM 256
#define BN 128
#define BK 32
#define LDSS 40   // padded row stride (f16 elems) to dodge bank conflicts

#if defined(__gfx1250__) && __has_builtin(__builtin_amdgcn_global_load_async_to_lds_b128)
#define HAS_ASYNC_LDS 1
typedef __attribute__((address_space(1))) v4i* as1_v4i;
typedef __attribute__((address_space(3))) v4i* as3_v4i;
#else
#define HAS_ASYNC_LDS 0
#endif

// C[M x N] (f32 accum, OutT out) = A[M x K] * B[K x N], all row-major.
// Batch z decomposed: zo = z / innerCount, zi = z % innerCount;
// per-matrix offset = zo*s?o + zi*s?i. M must be a multiple of 256, N of 128,
// K of 32 — true for every call site here.
template <typename AT, typename OutT>
__global__ __launch_bounds__(256)
void gemm_wmma(const AT* __restrict__ Ag, long long lda, long long sAo, long long sAi,
               const _Float16* __restrict__ Bg, long long ldb, long long sBo, long long sBi,
               OutT* __restrict__ Cg, long long ldc, long long sCo, long long sCi,
               int innerCount, int K)
{
    __shared__ _Float16 sA [2][BM * LDSS];   // row-major M x K tile (double buffered)
    __shared__ _Float16 sBt[2][BN * LDSS];   // TRANSPOSED: N x K tile (double buffered)

    const int z  = blockIdx.z;
    const int zo = z / innerCount;
    const int zi = z - zo * innerCount;
    const AT*       A = Ag + (size_t)(zo * sAo + zi * sAi);
    const _Float16* B = Bg + (size_t)(zo * sBo + zi * sBi);
    OutT*           C = Cg + (size_t)(zo * sCo + zi * sCi);

    const int blockRow = blockIdx.y * BM;
    const int blockCol = blockIdx.x * BN;

    const int t     = threadIdx.x;
    const int lane  = t & 31;
    const int wave  = t >> 5;
    const int wm    = wave & 3;     // 4 waves tall -> 64 rows each
    const int wn    = wave >> 2;    // 2 waves wide -> 64 cols each
    const int lhalf = lane >> 4;    // 0/1
    const int l16   = lane & 15;

    const v8f vzero = {0.f,0.f,0.f,0.f,0.f,0.f,0.f,0.f};
    v8f acc[4][4];
#pragma unroll
    for (int i = 0; i < 4; ++i)
#pragma unroll
        for (int j = 0; j < 4; ++j) acc[i][j] = vzero;

    // staging coordinates (256 threads)
    const int ar0  = t >> 2;          // 0..63  (x4 passes -> 256 rows)
    const int acol = (t & 3) * 8;     // 0,8,16,24
    const int bkr0 = t >> 4;          // 0..15  (x2 passes -> 32 k-rows)
    const int bcol = (t & 15) * 8;    // 0..120

    auto stageA = [&](int buf, int k0) {
#pragma unroll
        for (int i = 0; i < 4; ++i) {
            const int r = ar0 + i * 64;
            const AT* src = A + (size_t)(blockRow + r) * lda + k0 + acol;
            _Float16* dst = &sA[buf][r * LDSS + acol];
            if constexpr (sizeof(AT) == 2) {
#if HAS_ASYNC_LDS
                __builtin_amdgcn_global_load_async_to_lds_b128(
                    (as1_v4i)src, (as3_v4i)dst, 0, 0);
#else
                *(v8h*)dst = *(const v8h*)src;
#endif
            } else {
                v4f f0 = *(const v4f*)src;
                v4f f1 = *(const v4f*)(src + 4);
                v8h h;
#pragma unroll
                for (int e = 0; e < 4; ++e) { h[e] = (_Float16)f0[e]; h[e+4] = (_Float16)f1[e]; }
                *(v8h*)dst = h;
            }
        }
    };
    auto stageB = [&](int buf, int k0) {
#pragma unroll
        for (int i = 0; i < 2; ++i) {
            const int kr = bkr0 + i * 16;
            v8h hv = *(const v8h*)(B + (size_t)(k0 + kr) * ldb + blockCol + bcol);
#pragma unroll
            for (int e = 0; e < 8; ++e) sBt[buf][(bcol + e) * LDSS + kr] = hv[e];
        }
    };

    stageA(0, 0);
    stageB(0, 0);
    const int nT = K / BK;
    for (int it = 0; it < nT; ++it) {
        const int cur = it & 1;
#if HAS_ASYNC_LDS
        if constexpr (sizeof(AT) == 2)
            asm volatile("s_wait_asynccnt 0x0" ::: "memory");
#endif
        __syncthreads();
        if (it + 1 < nT) {              // overlap next-tile staging with WMMAs
            stageA(cur ^ 1, (it + 1) * BK);
            stageB(cur ^ 1, (it + 1) * BK);
        }

        const _Float16* Ab = &sA[cur][0];
        const _Float16* Bb = &sBt[cur][0];

        // A fragments: ISA 16-bit A 16x32 layout.
        // lane<16 : row=l16, K {kb..kb+7, kb+16..kb+23}, kb=0; lane>=16: kb=8.
        v16h af[4];
#pragma unroll
        for (int mt = 0; mt < 4; ++mt) {
            const int r  = wm * 64 + mt * 16 + l16;
            const int kb = lhalf * 8;
            const _Float16* p = Ab + r * LDSS + kb;
            v8h lo = *(const v8h*)p;
            v8h hi = *(const v8h*)(p + 16);
            v16h a;
#pragma unroll
            for (int e = 0; e < 8; ++e) { a[e] = lo[e]; a[e+8] = hi[e]; }
            af[mt] = a;
        }
        // B fragments: col l16, K {kb..kb+7, kb+16..kb+23}, kb = lhalf*8.
#pragma unroll
        for (int nt = 0; nt < 4; ++nt) {
            const int c  = wn * 64 + nt * 16 + l16;
            const int kb = lhalf * 8;
            const _Float16* p = Bb + c * LDSS + kb;
            v8h lo = *(const v8h*)p;
            v8h hi = *(const v8h*)(p + 16);
            v16h b;
#pragma unroll
            for (int e = 0; e < 8; ++e) { b[e] = lo[e]; b[e+8] = hi[e]; }
#pragma unroll
            for (int mt = 0; mt < 4; ++mt)
                acc[mt][nt] = __builtin_amdgcn_wmma_f32_16x16x32_f16(
                    false, af[mt], false, b, (short)0, acc[mt][nt], false, false);
        }
    }

    // writeout: C/D layout VGPR r, lane -> (M = r + 8*lhalf, N = l16)
#pragma unroll
    for (int mt = 0; mt < 4; ++mt)
#pragma unroll
        for (int nt = 0; nt < 4; ++nt) {
            const int rowBase = blockRow + wm * 64 + mt * 16 + lhalf * 8;
            const int col     = blockCol + wn * 64 + nt * 16 + l16;
#pragma unroll
            for (int r = 0; r < 8; ++r)
                C[(size_t)(rowBase + r) * ldc + col] = (OutT)acc[mt][nt][r];
        }
}

// Column-wise softmax (reference's axis=-2 quirk) over T=1024 rows, in place,
// on f32 scores. One block per (column, batch); logits scaled by 1/sqrt(T).
__global__ __launch_bounds__(256)
void col_softmax(float* __restrict__ S, int T, float scale)
{
    __shared__ float red[256];
    float* base = S + (size_t)blockIdx.y * T * T;
    const int col = blockIdx.x;
    const int t   = threadIdx.x;

    float v[4];
    float m = -3.4e38f;
#pragma unroll
    for (int i = 0; i < 4; ++i) {
        v[i] = base[(size_t)(t + i * 256) * T + col] * scale;
        m = fmaxf(m, v[i]);
    }
    red[t] = m; __syncthreads();
    for (int s = 128; s > 0; s >>= 1) { if (t < s) red[t] = fmaxf(red[t], red[t + s]); __syncthreads(); }
    m = red[0]; __syncthreads();

    float sum = 0.f;
#pragma unroll
    for (int i = 0; i < 4; ++i) { v[i] = __expf(v[i] - m); sum += v[i]; }
    red[t] = sum; __syncthreads();
    for (int s = 128; s > 0; s >>= 1) { if (t < s) red[t] += red[t + s]; __syncthreads(); }
    const float inv = 1.f / red[0];
#pragma unroll
    for (int i = 0; i < 4; ++i)
        base[(size_t)(t + i * 256) * T + col] = v[i] * inv;
}

__global__ void f32_to_f16(const float* __restrict__ in, _Float16* __restrict__ out, size_t n)
{
    size_t i = (size_t)blockIdx.x * blockDim.x + threadIdx.x;
    const size_t stride = (size_t)gridDim.x * blockDim.x;
    for (; i < n; i += stride) out[i] = (_Float16)in[i];
}

extern "C" void kernel_launch(void* const* d_in, const int* in_sizes, int n_in,
                              void* d_out, int out_size, void* d_ws, size_t ws_size,
                              hipStream_t stream)
{
    (void)in_sizes; (void)n_in; (void)out_size; (void)ws_size;
    const long long B = 8, T = 1024, E = 768, H = 12;

    const float* x  = (const float*)d_in[0];
    const float* WQ = (const float*)d_in[1];
    const float* WK = (const float*)d_in[2];
    const float* WV = (const float*)d_in[3];
    const float* WO = (const float*)d_in[4];
    float* out = (float*)d_out;

    char* ws = (char*)d_ws;
    size_t off = 0;
    auto carve = [&](size_t bytes) -> char* {
        char* p = ws + off;
        off += (bytes + 255) & ~(size_t)255;
        return p;
    };
    _Float16* x16  = (_Float16*)carve((size_t)(B*T*E)   * 2);
    _Float16* wq16 = (_Float16*)carve((size_t)(H*E*E)   * 2);
    _Float16* wk16 = (_Float16*)carve((size_t)(H*E*E)   * 2);
    _Float16* wv16 = (_Float16*)carve((size_t)(H*E*E)   * 2);
    _Float16* wo16 = (_Float16*)carve((size_t)(H*E*E)   * 2);   // H*E x E == 9216x768
    _Float16* q16  = (_Float16*)carve((size_t)(H*B*T*E) * 2);   // (h, b*t, e)
    _Float16* k16  = (_Float16*)carve((size_t)(H*B*T*E) * 2);   // (h, b*t, e)
    _Float16* v16  = (_Float16*)carve((size_t)(H*B*T*E) * 2);   // (h, b*t, e)
    float*    s32  = (float*)   carve((size_t)(H*B*T*T) * 4);   // (h, b, t, t)
    _Float16* z16  = (_Float16*)carve((size_t)(B*H*T*E) * 2);   // (b, h, t, e)  <- concat trick

    // ---- fp32 -> fp16 conversions ----
    f32_to_f16<<<2048, 256, 0, stream>>>(x,  x16, (size_t)(B*T*E));
    f32_to_f16<<<2048, 256, 0, stream>>>(WQ, wq16, (size_t)(H*E*E));
    f32_to_f16<<<2048, 256, 0, stream>>>(WK, wk16, (size_t)(H*E*E));
    f32_to_f16<<<2048, 256, 0, stream>>>(WV, wv16, (size_t)(H*E*E));
    f32_to_f16<<<2048, 256, 0, stream>>>(WO, wo16, (size_t)(H*E*E));

    // ---- Q/K/V projections: per-head GEMM, M=B*T=8192, N=K=E=768 ----
    {
        dim3 g((unsigned)(E/BN), (unsigned)(B*T/BM), (unsigned)H);
        gemm_wmma<_Float16,_Float16><<<g, 256, 0, stream>>>(
            x16, E, 0, 0,  wq16, E, E*E, 0,  q16, E, B*T*E, 0, 1, (int)E);
        gemm_wmma<_Float16,_Float16><<<g, 256, 0, stream>>>(
            x16, E, 0, 0,  wk16, E, E*E, 0,  k16, E, B*T*E, 0, 1, (int)E);
        gemm_wmma<_Float16,_Float16><<<g, 256, 0, stream>>>(
            x16, E, 0, 0,  wv16, E, E*E, 0,  v16, E, B*T*E, 0, 1, (int)E);
    }

    // ---- scores = Q @ reshape(K, (e, t)): per (h,b), M=T, N=T, K=E ----
    // K reshape quirk is free: the (t,e) K block reinterpreted row-major as
    // (e, t) is the same buffer with ldb = T.
    {
        dim3 g((unsigned)(T/BN), (unsigned)(T/BM), (unsigned)(H*B));
        gemm_wmma<_Float16,float><<<g, 256, 0, stream>>>(
            q16, E, B*T*E, T*E,
            k16, T, B*T*E, T*E,
            s32, T, B*T*T, T*T,
            (int)B, (int)E);
    }

    // ---- softmax over the QUERY axis (column-wise), scale 1/sqrt(T)=1/32 ----
    col_softmax<<<dim3((unsigned)T, (unsigned)(H*B)), 256, 0, stream>>>(s32, (int)T, 1.0f/32.0f);

    // ---- Z = A @ V: per (h,b), M=T, N=E, K=T; A is f32 (converted in-kernel).
    // Store Z as (b, h, t, e) so the reference's flat (h,t,dv)->(t,h*dv)
    // reshape becomes a plain reinterpretation.
    {
        dim3 g((unsigned)(E/BN), (unsigned)(T/BM), (unsigned)(H*B));
        gemm_wmma<float,_Float16><<<g, 256, 0, stream>>>(
            s32, T, B*T*T, T*T,
            v16, E, B*T*E, T*E,
            z16, E, T*E,   H*T*E,   // offset = h*(T*E) + b*(H*T*E)
            (int)B, (int)T);
    }

    // ---- out = Zc @ WO: per batch, M=T, N=E, K=H*E=9216; Zc is z16[b]
    // viewed row-major as (T, H*E).
    {
        dim3 g((unsigned)(E/BN), (unsigned)(T/BM), (unsigned)B);
        gemm_wmma<_Float16,float><<<g, 256, 0, stream>>>(
            z16, H*E, H*T*E, 0,
            wo16, E, 0, 0,
            out, E, T*E, 0,
            1, (int)(H*E));
    }
}
